// TreeFFNSeq2SeqBlock_45981919871642
// MI455X (gfx1250) — compile-verified, matrix operand order checked by reading, not run on hardware
//
#include <hip/hip_runtime.h>

#define BATCH   4
#define S_LEN   8192
#define DMODEL  512
#define NROWS   (BATCH * S_LEN)

typedef __attribute__((ext_vector_type(16))) __bf16        v16bf;
typedef __attribute__((ext_vector_type(8)))  float         v8f;
typedef __attribute__((ext_vector_type(4)))  unsigned int  vu4;
typedef __attribute__((ext_vector_type(2)))  float         vf2;
typedef __attribute__((ext_vector_type(4)))  float         vf4;

__device__ __forceinline__ unsigned short f32_to_bf16(float f) {
    unsigned u = __float_as_uint(f);
    unsigned r = 0x7FFFu + ((u >> 16) & 1u);
    return (unsigned short)((u + r) >> 16);
}

// ---- CDNA5 async global->LDS copy (ASYNCcnt path), 16B per lane -----------
__device__ __forceinline__ void async_ld_b128(unsigned lds_off, const void* g) {
    asm volatile("global_load_async_to_lds_b128 %0, %1, off"
                 :: "v"(lds_off), "v"((unsigned long long)(size_t)g)
                 : "memory");
}

// ---- WMMA helpers ----------------------------------------------------------
__device__ __forceinline__ v8f wmma_bf16(v16bf a, v16bf b, v8f c) {
    return __builtin_amdgcn_wmma_f32_16x16x32_bf16(
        /*neg_a=*/false, a, /*neg_b=*/false, b,
        /*c_mod=*/(short)0, c, /*reuse_a=*/false, /*reuse_b=*/false);
}

union FragU { vu4 q[2]; v16bf v; };

__device__ __forceinline__ v16bf load_frag(const unsigned short* p) {
    FragU u;
    const vu4* s = (const vu4*)p;
    u.q[0] = s[0];
    u.q[1] = s[1];
    return u.v;
}

// ---------------------------------------------------------------------------
// Weight pre-swizzle: f32 (K x 512 row-major) -> bf16 fragment-ordered layout
// out[kchunk][nblock(32)][lane(32)][pos(16)]
//   n       = nblock*16 + (lane & 15)
//   k_local = (pos&7) + 8*(lane>>4) + 16*(pos>>3)
// ---------------------------------------------------------------------------
__global__ void swz_kernel(const float* __restrict__ W,
                           unsigned short* __restrict__ out, int kchunks) {
    int idx = blockIdx.x * 256 + threadIdx.x;
    int total = kchunks * 32 * 512;
    if (idx >= total) return;
    int pos  = idx & 15;
    int lane = (idx >> 4) & 31;
    int nblk = (idx >> 9) & 31;
    int kc   = idx >> 14;
    int n  = nblk * 16 + (lane & 15);
    int kl = (pos & 7) + 8 * (lane >> 4) + 16 * (pos >> 3);
    int k  = kc * 32 + kl;
    out[idx] = f32_to_bf16(W[k * DMODEL + n]);
}

// ---------------------------------------------------------------------------
// LayerNorm: one 256-thread block per row of 512; writes f32 + bf16 copies
// ---------------------------------------------------------------------------
__global__ void __launch_bounds__(256)
ln_kernel(const float* __restrict__ in, const float* __restrict__ g,
          const float* __restrict__ b, float* __restrict__ of,
          unsigned short* __restrict__ ob) {
    int row = blockIdx.x;
    int tid = threadIdx.x;
    vf2 v = ((const vf2*)(in + (size_t)row * DMODEL))[tid];
    float s = v.x + v.y;
    float q = v.x * v.x + v.y * v.y;
    #pragma unroll
    for (int o = 16; o > 0; o >>= 1) {
        s += __shfl_down(s, o);
        q += __shfl_down(q, o);
    }
    __shared__ float ss[8], qq[8];
    if ((tid & 31) == 0) { ss[tid >> 5] = s; qq[tid >> 5] = q; }
    __syncthreads();
    s = 0.f; q = 0.f;
    #pragma unroll
    for (int i = 0; i < 8; ++i) { s += ss[i]; q += qq[i]; }
    float mu  = s * (1.0f / DMODEL);
    float var = q * (1.0f / DMODEL) - mu * mu;
    float rs  = rsqrtf(var + 1e-5f);
    float g0 = g[2 * tid], g1 = g[2 * tid + 1];
    float b0 = b[2 * tid], b1 = b[2 * tid + 1];
    float o0 = (v.x - mu) * rs * g0 + b0;
    float o1 = (v.y - mu) * rs * g1 + b1;
    vf2 ov; ov.x = o0; ov.y = o1;
    ((vf2*)(of + (size_t)row * DMODEL))[tid] = ov;
    unsigned h0 = f32_to_bf16(o0), h1 = f32_to_bf16(o1);
    ((unsigned*)ob)[row * (DMODEL / 2) + tid] = h0 | (h1 << 16);
}

// ---------------------------------------------------------------------------
// Stage one K-chunk (32 k-values) into one LDS buffer: 8 async b128 / thread.
// sbase = LDS byte base of the target buffer (32KB region):
//   [0,4K)=hs  [4K,8K)=hd  [8K,16K)=We  [16K,24K)=WgT  [24K,32K)=WgB
// ---------------------------------------------------------------------------
__device__ __forceinline__ void stage_chunk(unsigned sbase, unsigned aoffB,
                                            const unsigned short* hsG,
                                            const unsigned short* hdG,
                                            const unsigned short* WeSW,
                                            const unsigned short* WgSW,
                                            int nb0, int tid, int kc) {
    async_ld_b128(sbase + aoffB,        hsG + kc * 32);
    async_ld_b128(sbase + 4096 + aoffB, hdG + kc * 32);
    const unsigned short* weG = WeSW + (kc * 32 + nb0) * 512 + tid * 16;
    async_ld_b128(sbase + 8192  + tid * 32,      weG);
    async_ld_b128(sbase + 8192  + tid * 32 + 16, weG + 8);
    const unsigned short* gtG = WgSW + (kc * 32 + nb0) * 512 + tid * 16;
    async_ld_b128(sbase + 16384 + tid * 32,      gtG);
    async_ld_b128(sbase + 16384 + tid * 32 + 16, gtG + 8);
    const unsigned short* gbG = WgSW + ((kc + 16) * 32 + nb0) * 512 + tid * 16;
    async_ld_b128(sbase + 24576 + tid * 32,      gbG);
    async_ld_b128(sbase + 24576 + tid * 32 + 16, gbG + 8);
}

// ---------------------------------------------------------------------------
// One tree-FFN iteration (dense form):
//   for every row t: delta = sigmoid(hs.WgT + hd.WgB + bg) * relu(hs.We + be)
//   hout[t] = h[t] + (boundary-valid ? delta : 0)
// hs = h[t + sdelta] (sdelta = -1 enc, +1 dec), hd = h[t]
// Tile: 64 rows x 128 cols per WG; 8 waves each 16x64; K-chunks of 32.
// Double-buffered LDS; async copies overlapped with the WMMA burst.
// ---------------------------------------------------------------------------
__global__ void __launch_bounds__(256)
ffn_iter_kernel(const float* __restrict__ h_f32,
                const unsigned short* __restrict__ h_bf16,
                const unsigned short* __restrict__ WeSW,   // [16][32][512]
                const unsigned short* __restrict__ WgSW,   // [32][32][512]
                const float* __restrict__ be,
                const float* __restrict__ bg,
                float* __restrict__ ho_f32,
                unsigned short* __restrict__ ho_bf16,
                int sdelta) {
    __shared__ __align__(16) unsigned short smem[2 * 16384];   // 2 x 32KB

    const int tid  = threadIdx.x;
    const int lane = tid & 31;
    const int w    = tid >> 5;
    const int wm   = w & 3;    // wave row-block (16 rows each)
    const int wn   = w >> 2;   // wave col-half (64 cols each)

    const int tileM = blockIdx.x * 64;   // global row base
    const int nb0   = blockIdx.y * 8;    // global nblock base (128 cols)

    const v8f vz = {0.f, 0.f, 0.f, 0.f, 0.f, 0.f, 0.f, 0.f};
    v8f accm[4] = {vz, vz, vz, vz};
    v8f accg[4] = {vz, vz, vz, vz};

    // -- staging assignment (constant across K-chunks) --
    const int srow = tid >> 2;                       // 0..63 row within tile
    const int kg   = tid & 3;                        // 0..3 k-group of 8
    const int grow = tileM + srow;
    const int tcur = grow & (S_LEN - 1);
    const bool vsrc = ((unsigned)(tcur + sdelta)) < (unsigned)S_LEN;
    const int srcRow = grow + (vsrc ? sdelta : 0);
    const int llane  = (srow & 15) + 16 * (kg & 1);  // A-fragment swizzle
    const int mblk   = srow >> 4;
    const unsigned aoffB = (unsigned)((mblk * 512 + llane * 16 + 8 * (kg >> 1)) * 2);
    const unsigned smem0 = (unsigned)(size_t)&smem[0];

    const unsigned short* hsG = h_bf16 + srcRow * DMODEL + kg * 8;
    const unsigned short* hdG = h_bf16 + grow * DMODEL + kg * 8;

    // prologue: stage chunk 0 into buffer 0
    stage_chunk(smem0, aoffB, hsG, hdG, WeSW, WgSW, nb0, tid, 0);

    for (int kc = 0; kc < 16; ++kc) {
        const int par = kc & 1;
        if (kc + 1 < 16) {
            // stream next chunk into the other buffer behind the compute
            stage_chunk(smem0 + ((kc + 1) & 1) * 32768, aoffB,
                        hsG, hdG, WeSW, WgSW, nb0, tid, kc + 1);
            // async loads complete in order: <=8 outstanding => chunk kc landed
            asm volatile("s_wait_asynccnt 0x8" ::: "memory");
        } else {
            asm volatile("s_wait_asynccnt 0x0" ::: "memory");
        }
        __syncthreads();

        const unsigned short* sb = smem + par * 16384;

        // load A + all B fragments into distinct registers, then WMMA burst
        v16bf aS = load_frag(sb + 0    + wm * 512 + lane * 16);
        v16bf aD = load_frag(sb + 2048 + wm * 512 + lane * 16);
        v16bf bW[4], bT[4], bB[4];
        #pragma unroll
        for (int nb = 0; nb < 4; ++nb) {
            const int blk = wn * 4 + nb;
            bW[nb] = load_frag(sb + 4096  + blk * 512 + lane * 16);
            bT[nb] = load_frag(sb + 8192  + blk * 512 + lane * 16);
            bB[nb] = load_frag(sb + 12288 + blk * 512 + lane * 16);
        }
        #pragma unroll
        for (int nb = 0; nb < 4; ++nb) {
            accm[nb] = wmma_bf16(aS, bW[nb], accm[nb]);
            accg[nb] = wmma_bf16(aS, bT[nb], accg[nb]);
            accg[nb] = wmma_bf16(aD, bB[nb], accg[nb]);
        }
        __syncthreads();
    }

    // -- fused epilogue: bias, relu, sigmoid, boundary mask, residual --
    const int rlane = lane & 15;
    const int rhalf = lane >> 4;
    const int col0  = nb0 * 16 + wn * 64;
    #pragma unroll
    for (int nb = 0; nb < 4; ++nb) {
        const int col = col0 + nb * 16 + rlane;
        const float bev = be[col];
        const float bgv = bg[col];
        #pragma unroll
        for (int j = 0; j < 8; ++j) {
            const int row = tileM + wm * 16 + j + 8 * rhalf;
            const int tt  = row & (S_LEN - 1);
            const bool valid = ((unsigned)(tt + sdelta)) < (unsigned)S_LEN;
            float m = accm[nb][j] + bev;
            m = m > 0.f ? m : 0.f;
            float ga = accg[nb][j] + bgv;
            float gate = 1.f / (1.f + __expf(-ga));
            float delta = valid ? (gate * m) : 0.f;
            const int idx = row * DMODEL + col;
            float o = h_f32[idx] + delta;
            ho_f32[idx] = o;
            ho_bf16[idx] = f32_to_bf16(o);
        }
    }
}

// ---------------------------------------------------------------------------
__global__ void add_kernel(const float* __restrict__ a,
                           const float* __restrict__ b,
                           float* __restrict__ o, int n4) {
    int i = blockIdx.x * 256 + threadIdx.x;
    if (i < n4) {
        vf4 x = ((const vf4*)a)[i];
        vf4 y = ((const vf4*)b)[i];
        ((vf4*)o)[i] = x + y;
    }
}

// ---------------------------------------------------------------------------
extern "C" void kernel_launch(void* const* d_in, const int* in_sizes, int n_in,
                              void* d_out, int out_size, void* d_ws, size_t ws_size,
                              hipStream_t stream) {
    (void)in_sizes; (void)n_in; (void)out_size; (void)ws_size;
    const float* x        = (const float*)d_in[0];
    const float* ln_enc_g = (const float*)d_in[1];
    const float* ln_enc_b = (const float*)d_in[2];
    const float* ln_dec_g = (const float*)d_in[3];
    const float* ln_dec_b = (const float*)d_in[4];
    const float* enc_We   = (const float*)d_in[5];
    const float* enc_be   = (const float*)d_in[6];
    const float* enc_Wg   = (const float*)d_in[7];
    const float* enc_bg   = (const float*)d_in[8];
    const float* dec_We   = (const float*)d_in[9];
    const float* dec_be   = (const float*)d_in[10];
    const float* dec_Wg   = (const float*)d_in[11];
    const float* dec_bg   = (const float*)d_in[12];
    float* out = (float*)d_out;

    char* ws = (char*)d_ws;
    size_t o = 0;
    float* hA = (float*)(ws + o);            o += (size_t)NROWS * DMODEL * 4;
    float* hB = (float*)(ws + o);            o += (size_t)NROWS * DMODEL * 4;
    unsigned short* hA16 = (unsigned short*)(ws + o); o += (size_t)NROWS * DMODEL * 2;
    unsigned short* hB16 = (unsigned short*)(ws + o); o += (size_t)NROWS * DMODEL * 2;
    unsigned short* WeEnc = (unsigned short*)(ws + o); o += 512 * 512 * 2;
    unsigned short* WgEnc = (unsigned short*)(ws + o); o += 1024 * 512 * 2;
    unsigned short* WeDec = (unsigned short*)(ws + o); o += 512 * 512 * 2;
    unsigned short* WgDec = (unsigned short*)(ws + o); o += 1024 * 512 * 2;

    // weights -> bf16, fragment-ordered (deterministic, re-done every call)
    swz_kernel<<<16 * 64, 256, 0, stream>>>(enc_We, WeEnc, 16);
    swz_kernel<<<32 * 64, 256, 0, stream>>>(enc_Wg, WgEnc, 32);
    swz_kernel<<<16 * 64, 256, 0, stream>>>(dec_We, WeDec, 16);
    swz_kernel<<<32 * 64, 256, 0, stream>>>(dec_Wg, WgDec, 32);

    const dim3 fgrid(NROWS / 64, DMODEL / 128);
    const int n4 = NROWS * DMODEL / 4;

    // encoder: h_enc = LN(x); 3 iterations
    ln_kernel<<<NROWS, 256, 0, stream>>>(x, ln_enc_g, ln_enc_b, hA, hA16);
    for (int it = 0; it < 3; ++it) {
        ffn_iter_kernel<<<fgrid, 256, 0, stream>>>(hA, hA16, WeEnc, WgEnc,
                                                   enc_be, enc_bg, hB, hB16, -1);
        { float* t = hA; hA = hB; hB = t; }
        { unsigned short* t = hA16; hA16 = hB16; hB16 = t; }
    }
    // h = x + enc_out  (stored in d_out)
    add_kernel<<<n4 / 256, 256, 0, stream>>>(x, hA, out, n4);

    // decoder: h_dec = LN(h); 3 iterations (src = t+1)
    ln_kernel<<<NROWS, 256, 0, stream>>>(out, ln_dec_g, ln_dec_b, hA, hA16);
    for (int it = 0; it < 3; ++it) {
        ffn_iter_kernel<<<fgrid, 256, 0, stream>>>(hA, hA16, WeDec, WgDec,
                                                   dec_be, dec_bg, hB, hB16, +1);
        { float* t = hA; hA = hB; hB = t; }
        { unsigned short* t = hA16; hA16 = hB16; hB16 = t; }
    }
    // out = h + dec_out
    add_kernel<<<n4 / 256, 256, 0, stream>>>(out, hA, out, n4);
}
